// RNNDecoder_6554120094131
// MI455X (gfx1250) — matrix operand order repaired and checked by reading, not compiled, not run on hardware
//
#include <hip/hip_runtime.h>
#include <math.h>

// ---------------------------------------------------------------------------
// Problem constants (match the JAX reference)
// ---------------------------------------------------------------------------
#define SEQ_LEN 256
#define BATCH   256
#define HID     512
#define G3      1536      // 3*HID
#define LAT     128
#define OUTF    64
#define LAYERS  4

// WMMA fragment-ready weight tiling:
//   B-matrix tile = 16x16x(K=32) slice: 32 lanes x 16 bf16 = 1024 bytes
//   per matrix: NT=96 column tiles (1536/16) x KT=16 k tiles (512/32)
#define NT 96
#define KT 16
#define MAT_ELEMS (NT * KT * 32 * 16)   // 786432 bf16 = 1.5 MB

typedef __attribute__((ext_vector_type(16))) __bf16 v16bf;
typedef __attribute__((ext_vector_type(8)))  __bf16 v8bf;
typedef __attribute__((ext_vector_type(8)))  float  v8f;

union FragU { v16bf v; v8bf h[2]; };

__device__ __forceinline__ float sigmoid_f(float x) {
    return 1.0f / (1.0f + __expf(-x));
}
__device__ __forceinline__ float tanh_f(float x) {
    x = fminf(15.0f, fmaxf(-15.0f, x));
    float e = __expf(2.0f * x);
    return (e - 1.0f) / (e + 1.0f);
}

// ---------------------------------------------------------------------------
// Kernel 1: convert f32 GRU weights -> bf16 in WMMA-fragment-ready order.
// Source W is [3H][H] row major (n,k). As the WMMA B operand we need
// B[k][n] tiles; element (n,k): lane = (n%16) + (k%32 >= 16 ? 16 : 0),
// elem = k%16, tile = (n/16)*KT + k/32, 16 bf16 per lane, lane-major.
// ---------------------------------------------------------------------------
__global__ void k_convert_weights(const float* __restrict__ whh,
                                  const float* __restrict__ wihr,
                                  __bf16* __restrict__ whhF,
                                  __bf16* __restrict__ wihF) {
    const int tot_hh = LAYERS * G3 * HID;
    const int tot_ih = (LAYERS - 1) * G3 * HID;
    for (int i = blockIdx.x * blockDim.x + threadIdx.x;
         i < tot_hh + tot_ih; i += gridDim.x * blockDim.x) {
        const float* src;
        __bf16* dst;
        int rem;
        if (i < tot_hh) { src = whh;  dst = whhF; rem = i; }
        else            { src = wihr; dst = wihF; rem = i - tot_hh; }
        int l  = rem / (G3 * HID);
        int r2 = rem % (G3 * HID);
        int n  = r2 / HID;
        int k  = r2 % HID;
        int nt = n >> 4, nl = n & 15;
        int kt = k >> 5, kr = k & 31;
        int lane = nl + (kr & 16);
        int elem = kr & 15;
        long d = (long)l * MAT_ELEMS +
                 (((long)(nt * KT + kt)) * 32 + lane) * 16 + elem;
        dst[d] = (__bf16)src[rem];
    }
}

// ---------------------------------------------------------------------------
// Kernel 2: layer-0 input gates, constant across time:
//   xg0[b][n] = x[b][:] . w_ih0[n][:] + b_ih[0][n]      (f32, 50 MFLOP)
// ---------------------------------------------------------------------------
__global__ void k_xg0(const float* __restrict__ x,
                      const float* __restrict__ wih0,
                      const float* __restrict__ bih0,
                      float* __restrict__ xg0) {
    int idx = blockIdx.x * blockDim.x + threadIdx.x;
    if (idx >= BATCH * G3) return;
    int b = idx / G3, n = idx % G3;
    const float* xr = x + b * LAT;
    const float* wr = wih0 + n * LAT;
    float acc = bih0[n];
    #pragma unroll 4
    for (int k = 0; k < LAT; k++) acc += xr[k] * wr[k];
    xg0[idx] = acc;
}

// ---------------------------------------------------------------------------
// Kernel 3: persistent GRU layer. 64 WGs x 256 threads (8 wave32).
//   WG (rowStrip 0..3, colg 0..15): 64 batch rows x 32 h-columns.
//   Wave w: row tile rt=w&3, col half c=w>>2 -> all 3 gates of one 16-col tile.
//   LDS holds this WG's W_hh (and W_ih) column slice, loaded once.
//   The f32 hidden state lives in registers (tile ownership is
//   time-invariant); the bf16 h for the A-matrix IS ys[t-1], so the only
//   per-step stores are the 8 b16 ys elements per lane.
// ---------------------------------------------------------------------------
__global__ void __launch_bounds__(256)
k_gru_layer(const __bf16* __restrict__ whhF,   // [NT][KT] tiles, this layer
            const __bf16* __restrict__ wihF,   // same (layers>0)
            const float*  __restrict__ xg0,    // [B][G3]  (layer 0 only)
            const float*  __restrict__ bih,    // b_ih[l] (G3) (layers>0)
            const float*  __restrict__ bhh,    // b_hh[l] (G3)
            const __bf16* __restrict__ ysPrev, // [T][B][H] (layers>0)
            __bf16*       __restrict__ ysOut,  // [T][B][H]
            const __bf16* __restrict__ zeroH,  // [B][H] zeros (t=0 state)
            unsigned* barCount, unsigned* barSense,
            int layer0) {
    extern __shared__ char lds[];

    const int wg       = blockIdx.x;       // 0..63
    const int rowBase  = (wg >> 4) * 64;   // 4 strips of 64 batch rows
    const int colg     = wg & 15;          // 16 groups of 32 h-columns
    const int tid      = threadIdx.x;
    const int wave     = tid >> 5;
    const int lane     = tid & 31;
    const int nm       = layer0 ? 1 : 2;   // matrices: W_hh (+ W_ih)

    // ---- stage weight slice into LDS (once) --------------------------------
    // lc = c*3 + g -> global column tile nt = g*32 + colg*2 + c.
    for (int m = 0; m < nm; m++) {
        const __bf16* base = (m == 0) ? whhF : wihF;
        for (int lc = 0; lc < 6; lc++) {
            int cc = lc / 3, g = lc % 3;
            int nt = g * 32 + colg * 2 + cc;
            const uint4* src = (const uint4*)(base + (long)nt * KT * 512);
            uint4* dst = (uint4*)(lds + (m * 6 + lc) * 16384);
            #pragma unroll
            for (int i = tid; i < 1024; i += 256) dst[i] = src[i];
        }
    }
    __syncthreads();

    const int rt = wave & 3;            // row tile within strip
    const int c  = wave >> 2;           // column half (0/1)
    const int colBaseH = colg * 32 + c * 16;          // h column of tile
    const int j   = colBaseH + (lane & 15);           // this lane's h column
    const int rA  = rowBase + rt * 16 + (lane & 15);  // A-fragment row
    const int k0off = (lane & 16);                    // 0 / 16 (A lane K split)
    const int mOff  = (lane & 16) >> 1;               // +8 rows for lanes 16..31
    const int rEp0  = rowBase + rt * 16 + mOff;       // epilogue row for i=0

    // ---- time-invariant per-lane state hoisted into registers --------------
    v8f hreg;                      // f32 hidden state, carried across steps
    #pragma unroll
    for (int i = 0; i < 8; i++) hreg[i] = 0.0f;

    float bh[3], bi[3];
    v8f xg0reg[3];
    #pragma unroll
    for (int g = 0; g < 3; g++) {
        bh[g] = bhh[g * HID + j];
        bi[g] = layer0 ? 0.0f : bih[g * HID + j];
        if (layer0) {
            #pragma unroll
            for (int i = 0; i < 8; i++)
                xg0reg[g][i] = xg0[(long)(rEp0 + i) * G3 + g * HID + j];
        }
    }

    for (int t = 0; t < SEQ_LEN; t++) {
        // h(t-1) as bf16 A-matrix is simply ys[t-1] (zeros at t=0)
        const __bf16* hIn = (t == 0) ? zeroH
                                     : (ysOut + (long)(t - 1) * BATCH * HID);

        // ---- accumulator init ----------------------------------------------
        v8f accH[3], accX[3];
        #pragma unroll
        for (int g = 0; g < 3; g++) {
            #pragma unroll
            for (int i = 0; i < 8; i++) accH[g][i] = bh[g];
            if (layer0) accX[g] = xg0reg[g];
            else {
                #pragma unroll
                for (int i = 0; i < 8; i++) accX[g][i] = bi[g];
            }
        }

        // ---- WMMA K loops: m=0: h @ W_hh^T -> accH ; m=1: ys @ W_ih^T -> accX
        #pragma unroll
        for (int m = 0; m < 2; m++) {
            if (m >= nm) break;
            const __bf16* A = (m == 0) ? hIn
                                       : (ysPrev + (long)t * BATCH * HID);
            const __bf16* arow = A + (long)rA * HID;
            // prefetch next timestep's activation row into L2
            if (m != 0 && t + 1 < SEQ_LEN)
                __builtin_prefetch(ysPrev + ((long)(t + 1) * BATCH + rA) * HID, 0, 1);
            #pragma unroll 4
            for (int kt = 0; kt < KT; kt++) {
                FragU a;
                int kk = kt * 32 + k0off;
                a.h[0] = *(const v8bf*)(arow + kk);
                a.h[1] = *(const v8bf*)(arow + kk + 8);
                #pragma unroll
                for (int g = 0; g < 3; g++) {
                    const int lc = c * 3 + g;
                    FragU b;
                    const v8bf* bp = (const v8bf*)(lds +
                        (((m * 6 + lc) * KT + kt) * 1024) + lane * 32);
                    b.h[0] = bp[0];
                    b.h[1] = bp[1];
                    v8f& acc = (m == 0) ? accH[g] : accX[g];
                    acc = __builtin_amdgcn_wmma_f32_16x16x32_bf16(
                        false, a.v, false, b.v, (short)0, acc, false, false);
                }
            }
        }

        // ---- GRU epilogue (torch gate order r,z,n); h = (1-z)n + z h -------
        #pragma unroll
        for (int i = 0; i < 8; i++) {
            float rr = sigmoid_f(accX[0][i] + accH[0][i]);
            float zz = sigmoid_f(accX[1][i] + accH[1][i]);
            float nn = tanh_f(accX[2][i] + rr * accH[2][i]);
            float hnew = (1.0f - zz) * nn + zz * hreg[i];
            hreg[i] = hnew;
            ysOut[((long)t * BATCH + (rEp0 + i)) * HID + j] = (__bf16)hnew;
        }

        // ---- device-wide barrier between timesteps -------------------------
        __threadfence();
        __syncthreads();
        if (tid == 0) {
            unsigned gen = __hip_atomic_load(barSense, __ATOMIC_ACQUIRE,
                                             __HIP_MEMORY_SCOPE_AGENT);
            unsigned prev = __hip_atomic_fetch_add(barCount, 1u,
                                                   __ATOMIC_ACQ_REL,
                                                   __HIP_MEMORY_SCOPE_AGENT);
            if (prev == (unsigned)(gridDim.x - 1)) {
                __hip_atomic_store(barCount, 0u, __ATOMIC_RELAXED,
                                   __HIP_MEMORY_SCOPE_AGENT);
                __hip_atomic_fetch_add(barSense, 1u, __ATOMIC_ACQ_REL,
                                       __HIP_MEMORY_SCOPE_AGENT);
            } else {
                while (__hip_atomic_load(barSense, __ATOMIC_ACQUIRE,
                                         __HIP_MEMORY_SCOPE_AGENT) == gen) {
                    __builtin_amdgcn_s_sleep(1);
                }
            }
        }
        __syncthreads();
    }
}

// ---------------------------------------------------------------------------
// Kernel 4: final FC, note output layout transpose [B][T][OUTF] from [T][B][H].
// ---------------------------------------------------------------------------
__global__ void k_fc(const __bf16* __restrict__ ys,
                     const float* __restrict__ fcw,
                     const float* __restrict__ fcb,
                     float* __restrict__ out) {
    long idx = (long)blockIdx.x * blockDim.x + threadIdx.x;
    if (idx >= (long)BATCH * SEQ_LEN * OUTF) return;
    int o = idx & 63;
    int t = (int)((idx >> 6) & 255);
    int b = (int)(idx >> 14);
    const __bf16* yr = ys + ((long)t * BATCH + b) * HID;
    const float*  wr = fcw + o * HID;
    float acc = fcb[o];
    #pragma unroll 8
    for (int h = 0; h < HID; h++) acc += (float)yr[h] * wr[h];
    out[idx] = acc;
}

// ---------------------------------------------------------------------------
// Host launcher
// ---------------------------------------------------------------------------
extern "C" void kernel_launch(void* const* d_in, const int* in_sizes, int n_in,
                              void* d_out, int out_size, void* d_ws, size_t ws_size,
                              hipStream_t stream) {
    const float* x       = (const float*)d_in[0];
    const float* w_ih0   = (const float*)d_in[1];
    const float* w_ihr   = (const float*)d_in[2];
    const float* w_hh    = (const float*)d_in[3];
    const float* b_ih    = (const float*)d_in[4];
    const float* b_hh    = (const float*)d_in[5];
    const float* fc_w    = (const float*)d_in[6];
    const float* fc_b    = (const float*)d_in[7];
    float* out = (float*)d_out;

    // Workspace carve-up (256B aligned)
    char* w = (char*)d_ws;
    size_t off = 0;
    auto take = [&](size_t bytes) {
        char* p = w + off;
        off = (off + bytes + 255) & ~(size_t)255;
        return p;
    };
    __bf16*   whhF  = (__bf16*)take((size_t)LAYERS * MAT_ELEMS * 2);
    __bf16*   wihF  = (__bf16*)take((size_t)(LAYERS - 1) * MAT_ELEMS * 2);
    float*    xg0   = (float*)take((size_t)BATCH * G3 * 4);
    __bf16*   zeroH = (__bf16*)take((size_t)BATCH * HID * 2);
    __bf16*   ysA   = (__bf16*)take((size_t)SEQ_LEN * BATCH * HID * 2);
    __bf16*   ysB   = (__bf16*)take((size_t)SEQ_LEN * BATCH * HID * 2);
    unsigned* barCt = (unsigned*)take(256);
    unsigned* barSn = barCt + 1;
    (void)ws_size; (void)n_in; (void)in_sizes; (void)out_size;

    // Barrier state and the zero h(0) buffer must start at 0 every call.
    hipMemsetAsync(barCt, 0, 256, stream);
    hipMemsetAsync(zeroH, 0, (size_t)BATCH * HID * 2, stream);

    // 1) weight conversion to fragment-ready bf16
    {
        int tot = LAYERS * G3 * HID + (LAYERS - 1) * G3 * HID;
        int blocks = (tot + 255) / 256;
        k_convert_weights<<<blocks, 256, 0, stream>>>(w_hh, w_ihr, whhF, wihF);
    }
    // 2) layer-0 input gates (time-invariant)
    {
        int blocks = (BATCH * G3 + 255) / 256;
        k_xg0<<<blocks, 256, 0, stream>>>(x, w_ih0, b_ih, xg0);
    }
    // 3) GRU layers (persistent kernel per layer, ping-pong ys buffers)
    __bf16* ysPing[2] = { ysA, ysB };
    for (int l = 0; l < LAYERS; l++) {
        const __bf16* wh = whhF + (size_t)l * MAT_ELEMS;
        const __bf16* wi = (l > 0) ? (wihF + (size_t)(l - 1) * MAT_ELEMS) : whhF;
        const __bf16* ysIn  = ysPing[(l + 1) & 1];  // layer0: unused
        __bf16*       ysOut = ysPing[l & 1];        // l0->A, l1->B, l2->A, l3->B
        int layer0 = (l == 0) ? 1 : 0;
        size_t ldsBytes = layer0 ? (6u * KT * 1024u)        //  96 KB
                                 : (12u * KT * 1024u);      // 192 KB
        k_gru_layer<<<64, 256, ldsBytes, stream>>>(
            wh, wi, xg0, b_ih + (size_t)l * G3, b_hh + (size_t)l * G3,
            ysIn, ysOut, zeroH, barCt, barSn, layer0);
    }
    // 4) FC head (reads ys of layer 3 == ysB)
    {
        long tot = (long)BATCH * SEQ_LEN * OUTF;
        int blocks = (int)((tot + 255) / 256);
        k_fc<<<blocks, 256, 0, stream>>>(ysB, fc_w, fc_b, out);
    }
}
